// SimpleFC_57655640981545
// MI455X (gfx1250) — compile-verified, matrix operand order checked by reading, not run on hardware
//
#include <hip/hip_runtime.h>
#include <hip/hip_bf16.h>

// SimpleFC transformer for MI455X (gfx1250, wave32, WMMA + TDM).
// B=16, N=1024, E=64, H=4, dh=16. ~11 GFLOP, ~50MB traffic -> compute bound in
// the N^2 attention matmuls. All GEMMs run on v_wmma_f32_16x16x32_f16 (fp16
// operands, f32 accumulation); dh=16 is zero-padded to K=32. Weight tiles are
// DMA'd into LDS by the Tensor Data Mover where the toolchain exposes it.
// Attention computes S^T = K*Q^T so the softmax reduction lies along the VGPR
// axis (pure VALU) instead of the lane axis; softmax runs in log2 domain on
// raw v_exp_f32.

#define Bb 16
#define Nn 1024
#define Ee 64
#define Hh 4
#define ROWS (Bb * Nn)  // 16384 tokens

typedef __attribute__((ext_vector_type(16))) _Float16 v16h;
typedef __attribute__((ext_vector_type(8)))  _Float16 v8h;
typedef __attribute__((ext_vector_type(8)))  float    v8f;
typedef __attribute__((ext_vector_type(4)))  unsigned u32x4;
typedef __attribute__((ext_vector_type(8)))  int      i32x8;
typedef __attribute__((ext_vector_type(4)))  int      i32x4;

#if defined(__has_builtin)
#if __has_builtin(__builtin_amdgcn_tensor_load_to_lds) && __has_builtin(__builtin_amdgcn_s_wait_tensorcnt)
#define HAVE_TDM 1
#endif
#if __has_builtin(__builtin_amdgcn_exp2f)
#define fast_exp2(x) __builtin_amdgcn_exp2f(x)   // bare v_exp_f32; FTZ on
#endif
#endif
#ifndef HAVE_TDM
#define HAVE_TDM 0
#endif
#ifndef fast_exp2
#define fast_exp2(x) exp2f(x)
#endif

// 16x32 fp16 A/B fragment: each lane supplies two contiguous 16B runs
// (K 0..7 / 16..23 for lanes 0-15, K 8..15 / 24..31 for lanes 16-31).
__device__ __forceinline__ v16h make_frag(const _Float16* p0, const _Float16* p1) {
  v8h lo = *(const v8h*)p0;
  v8h hi = *(const v8h*)p1;
  v16h r;
#pragma unroll
  for (int i = 0; i < 8; ++i) { r[i] = lo[i]; r[i + 8] = hi[i]; }
  return r;
}

// ---------------------------------------------------------------------------
// Time embedding: pos_enc -> silu -> two 64x64 matvecs. grid=B, block=64.
__global__ void embed_kernel(const float* __restrict__ t_int,
                             const float* __restrict__ e1w, const float* __restrict__ e1b,
                             const float* __restrict__ e2w, const float* __restrict__ e2b,
                             float* __restrict__ emb1, float* __restrict__ emb2) {
  __shared__ float s[Ee];
  int b = blockIdx.x, e = threadIdx.x;
  float t = t_int[b];
  float ti;
  if (e < 32) { float f = __powf(10000.f, -(float)(2 * e) / 64.f);        ti = __sinf(t * f); }
  else        { float f = __powf(10000.f, -(float)(2 * (e - 32)) / 64.f); ti = __cosf(t * f); }
  s[e] = ti / (1.f + __expf(-ti));  // silu
  __syncthreads();
  float a1 = e1b[e], a2 = e2b[e];
#pragma unroll 8
  for (int k = 0; k < Ee; ++k) { a1 += s[k] * e1w[e * Ee + k]; a2 += s[k] * e2w[e * Ee + k]; }
  emb1[b * Ee + e] = a1;
  emb2[b * Ee + e] = a2;
}

// x = pos @ w_in^T + b_in + emb1[b].  K=3 -> plain VALU.
__global__ void input_proj_kernel(const float* __restrict__ pos,
                                  const float* __restrict__ w_in, const float* __restrict__ b_in,
                                  const float* __restrict__ emb1, float* __restrict__ x) {
  int idx = blockIdx.x * 256 + threadIdx.x;  // ROWS*64 total
  int t = idx >> 6, e = idx & 63, b = t >> 10;
  const float* pp = pos + t * 3;
  const float* wp = w_in + e * 3;
  x[idx] = b_in[e] + emb1[b * Ee + e] + pp[0] * wp[0] + pp[1] * wp[1] + pp[2] * wp[2];
}

// LayerNorm, one wave per token (lane handles e and e+32), fp16 output.
__global__ __launch_bounds__(128) void ln_kernel(const float* __restrict__ x,
                                                 const float* __restrict__ g,
                                                 const float* __restrict__ b,
                                                 _Float16* __restrict__ out) {
  int wave = threadIdx.x >> 5, lane = threadIdx.x & 31;
  int t = blockIdx.x * 4 + wave;
  const float* xp = x + (long)t * Ee;
  float x0 = xp[lane], x1 = xp[lane + 32];
  float s = x0 + x1;
#pragma unroll
  for (int m = 16; m >= 1; m >>= 1) s += __shfl_xor(s, m, 32);
  float mean = s * (1.f / 64.f);
  float d0 = x0 - mean, d1 = x1 - mean;
  float v = d0 * d0 + d1 * d1;
#pragma unroll
  for (int m = 16; m >= 1; m >>= 1) v += __shfl_xor(v, m, 32);
  float inv = rsqrtf(v * (1.f / 64.f) + 1e-5f);
  out[(long)t * Ee + lane]      = (_Float16)(d0 * inv * g[lane] + b[lane]);
  out[(long)t * Ee + lane + 32] = (_Float16)(d1 * inv * g[lane + 32] + b[lane + 32]);
}

// ---------------------------------------------------------------------------
// WMMA GEMM, epilogue specialized at compile time (branch-free j-loop).
template <int ACT_GELU, int BVEC, int RESID, int STOREF, int S16MODE, int QKV>
__global__ __launch_bounds__(256) void gemm_wmma_kernel(
    const _Float16* __restrict__ X, const float* __restrict__ W,
    const float* __restrict__ bias,
    float* __restrict__ Yf, _Float16* __restrict__ Y16,
    const float* __restrict__ resid, const float* __restrict__ bvec,
    _Float16* __restrict__ q16, _Float16* __restrict__ k16, _Float16* __restrict__ v16,
    int ncols) {
  __shared__ __align__(16) _Float16 Wl[192 * 64];
  int tid = threadIdx.x;
  int total = ncols * 64;

#if HAVE_TDM
  // Tensor Data Mover: DMA the f32 weight tile [ncols,64] global -> LDS stage,
  // then convert to fp16 cooperatively. D# per ISA 8.3/8.4: count=1, type=2,
  // data_size=4B, tile==tensor, dim0 stride 64.
  __shared__ __align__(16) float Wstage[192 * 64];
  if (tid < 32) {
    // low 32 bits of a generic LDS address are the LDS byte offset (ISA 10.2)
    unsigned lds_off = (unsigned)(unsigned long long)(void*)&Wstage[0];
    unsigned long long ga = (unsigned long long)W;
    u32x4 g0;
    g0[0] = 1u;                                   // count=1, user descriptor
    g0[1] = lds_off;                              // lds_addr
    g0[2] = (unsigned)ga;                         // global_addr[31:0]
    g0[3] = (unsigned)(ga >> 32) | 0x80000000u;   // global_addr[56:32] | type=2
    i32x8 g1;
    g1[0] = (2 << 16);                            // data_size = 4 bytes
    g1[1] = (64 << 16);                           // tensor_dim0 = 64
    g1[2] = (int)((unsigned)ncols << 16);         // tensor_dim1 = ncols
    g1[3] = (64 << 16);                           // tile_dim0 = 64
    g1[4] = ncols;                                // tile_dim1 = ncols
    g1[5] = 64;                                   // tensor_dim0_stride = 64
    g1[6] = 0;
    g1[7] = 0;
    i32x4 gz = {0, 0, 0, 0};
#if defined(__clang_major__) && __clang_major__ >= 23
    i32x8 gz8 = {0, 0, 0, 0, 0, 0, 0, 0};
    __builtin_amdgcn_tensor_load_to_lds(g0, g1, gz, gz, gz8, 0);
#else
    __builtin_amdgcn_tensor_load_to_lds(g0, g1, gz, gz, 0);
#endif
    __builtin_amdgcn_s_wait_tensorcnt(0);
  }
  __syncthreads();
  for (int i = tid; i < total; i += 256) Wl[i] = (_Float16)Wstage[i];
#else
  for (int i = tid; i < total; i += 256) Wl[i] = (_Float16)W[i];
#endif
  __syncthreads();

  int wave = tid >> 5, lane = tid & 31;
  int rowbase = blockIdx.x * 128 + wave * 16;
  int mrow = rowbase + (lane & 15);
  int khalf = (lane >> 4) * 8;
  const _Float16* xr = X + (long)mrow * 64;
  v16h a0 = make_frag(xr + 0  + khalf, xr + 16 + khalf);
  v16h a1 = make_frag(xr + 32 + khalf, xr + 48 + khalf);

  int nct = ncols >> 4;
  for (int ct = 0; ct < nct; ++ct) {
    int colbase = ct * 16;
    int cb = (colbase + (lane & 15)) * 64 + khalf;
    v16h b0 = make_frag(&Wl[cb],      &Wl[cb + 16]);
    v16h b1 = make_frag(&Wl[cb + 32], &Wl[cb + 48]);
    v8f c = {};
    c = __builtin_amdgcn_wmma_f32_16x16x32_f16(false, a0, false, b0, (short)0, c, false, false);
    c = __builtin_amdgcn_wmma_f32_16x16x32_f16(false, a1, false, b1, (short)0, c, false, false);
#pragma unroll
    for (int j = 0; j < 8; ++j) {
      int row = rowbase + j + ((lane >= 16) ? 8 : 0);
      int col = colbase + (lane & 15);
      float val = c[j] + bias[col];
      if constexpr (ACT_GELU) val = 0.5f * val * (1.f + erff(val * 0.70710678118f));
      if constexpr (BVEC)  val += bvec[(row >> 10) * Ee + col];
      if constexpr (RESID) val += resid[(long)row * Ee + col];
      if constexpr (STOREF) Yf[(long)row * Ee + col] = val;
      if constexpr (S16MODE == 1) Y16[(long)row * Ee + col] = (_Float16)val;
      if constexpr (S16MODE == 2) Y16[(long)row * Ee + col] = (_Float16)fmaxf(val, 0.f);
      if constexpr (QKV) {
        int part = col >> 6;           // 0=q 1=k 2=v
        int hh   = (col >> 4) & 3;
        int d    = col & 15;
        int b_ = row >> 10, n_ = row & 1023;
        long base = ((long)(b_ * Hh + hh) * Nn + n_);
        // fold 1/sqrt(dh) * log2(e) into Q so softmax runs on bare exp2
        if (part == 0)      q16[base * 32 + d] = (_Float16)(val * (0.25f * 1.44269504089f));
        else if (part == 1) k16[base * 32 + d] = (_Float16)val;
        else                v16[base * 16 + d] = (_Float16)val;
      }
    }
  }
}

// ---------------------------------------------------------------------------
// Flash attention, one wave per 16-query tile.
// S^T = K*Q^T via WMMA: C-layout puts the key index on the VGPR axis, so the
// per-query max/sum over 16 keys is 7 VALU ops + one xor-16 shuffle. The tiny
// C-orientation broadcasts (O rescale, final 1/l) go through a 16-float LDS
// row; the P tile is stored transposed into LDS, which simultaneously
// converts it to the A-layout needed by the PV WMMA.
__global__ __launch_bounds__(128) void attn_kernel(const _Float16* __restrict__ q16,
                                                   const _Float16* __restrict__ k16,
                                                   const _Float16* __restrict__ v16,
                                                   _Float16* __restrict__ o16) {
  __shared__ __align__(16) _Float16 Plds[4 * 16 * 32];
  __shared__ float Slds[4 * 16];
  int wave = threadIdx.x >> 5, lane = threadIdx.x & 31;
  _Float16* P = Plds + wave * 16 * 32;
  float* S = Slds + wave * 16;
  for (int i = lane; i < 16 * 16; i += 32) {  // zero the K=16..31 pad half once
    int r = i >> 4, cc = i & 15;
    P[r * 32 + 16 + cc] = (_Float16)0.f;
  }
  asm volatile("s_wait_dscnt 0x0" ::: "memory");

  int wg = blockIdx.x * 4 + wave;   // 4096 waves total
  int qt = wg & 63;                 // query tile within sequence
  int bh = wg >> 6;                 // b*H + h
  int khalf = (lane >> 4) * 8;
  int half8 = (lane >= 16) ? 8 : 0;
  int qcol  = lane & 15;            // this lane's query (after half-merge)
  long qrow = ((long)bh * Nn + qt * 16 + qcol);
  v16h qb = make_frag(q16 + qrow * 32 + khalf, q16 + qrow * 32 + 16 + khalf);  // B operand

  v8f o = {};
  float m = -INFINITY, lsum = 0.f;

  for (int kt = 0; kt < Nn / 16; ++kt) {
    long krow = ((long)bh * Nn + kt * 16 + qcol);
    v16h ka = make_frag(k16 + krow * 32 + khalf, k16 + krow * 32 + 16 + khalf);  // A operand
    if (kt + 1 < Nn / 16) __builtin_prefetch(k16 + (krow + 16) * 32, 0, 3);

    v8f st = {};  // S^T tile: element (key = j + half8, query = qcol)
    st = __builtin_amdgcn_wmma_f32_16x16x32_f16(false, ka, false, qb, (short)0, st, false, false);

    float tmax = st[0];
#pragma unroll
    for (int j = 1; j < 8; ++j) tmax = fmaxf(tmax, st[j]);
    tmax = fmaxf(tmax, __shfl_xor(tmax, 16, 32));   // merge the two key halves
    float nm = fmaxf(m, tmax);
    float sc = fast_exp2(m - nm);                    // log2-domain (Q pre-scaled)
    m = nm;

    float pv[8], psum = 0.f;
#pragma unroll
    for (int j = 0; j < 8; ++j) { pv[j] = fast_exp2(st[j] - nm); psum += pv[j]; }
    psum += __shfl_xor(psum, 16, 32);
    lsum = lsum * sc + psum;

    S[qcol] = sc;  // both halves write the same value: benign
#pragma unroll
    for (int j = 0; j < 8; ++j)
      P[qcol * 32 + j + half8] = (_Float16)pv[j];    // transposed store -> A-layout
    asm volatile("s_wait_dscnt 0x0" ::: "memory");

    v16h pa = make_frag(&P[qcol * 32 + khalf], &P[qcol * 32 + 16 + khalf]);
#pragma unroll
    for (int j = 0; j < 8; ++j) o[j] *= S[j + half8];  // broadcast ds_load_b32

    v16h vb;  // B frag of V: B[k][d] = V[kt*16+k][d]; K 16..31 zero (matches P pad)
    long vbase = ((long)bh * Nn + kt * 16 + khalf) * 16 + qcol;
#pragma unroll
    for (int j = 0; j < 8; ++j) { vb[j] = v16[vbase + j * 16]; vb[j + 8] = (_Float16)0.f; }

    o = __builtin_amdgcn_wmma_f32_16x16x32_f16(false, pa, false, vb, (short)0, o, false, false);
  }

  S[qcol] = 1.0f / lsum;  // LDS in-order per wave: lands after the loop's last reads
  asm volatile("s_wait_dscnt 0x0" ::: "memory");
  int b_ = bh >> 2, h_ = bh & 3;
#pragma unroll
  for (int j = 0; j < 8; ++j) {
    float rin = S[j + half8];
    int row = qt * 16 + j + half8;
    int col = h_ * 16 + qcol;
    o16[((long)b_ * Nn + row) * Ee + col] = (_Float16)(o[j] * rin);
  }
}

// Final head: relu(x) @ w_out^T + b_out, masked. 3 cols -> VALU.
__global__ void out_proj_kernel(const float* __restrict__ x,
                                const float* __restrict__ w_out, const float* __restrict__ b_out,
                                const float* __restrict__ node_mask, float* __restrict__ out) {
  int idx = blockIdx.x * 256 + threadIdx.x;
  if (idx >= ROWS * 3) return;
  int t = idx / 3, c = idx % 3;
  const float* xp = x + (long)t * Ee;
  const float* wp = w_out + c * Ee;
  float v = b_out[c];
#pragma unroll 8
  for (int k = 0; k < Ee; ++k) v += fmaxf(xp[k], 0.f) * wp[k];
  out[idx] = v * node_mask[t];
}

// ---------------------------------------------------------------------------
extern "C" void kernel_launch(void* const* d_in, const int* in_sizes, int n_in,
                              void* d_out, int out_size, void* d_ws, size_t ws_size,
                              hipStream_t stream) {
  const float* pos       = (const float*)d_in[0];
  const float* t_int     = (const float*)d_in[1];
  const float* node_mask = (const float*)d_in[2];
  const float* w_in  = (const float*)d_in[3];
  const float* b_in  = (const float*)d_in[4];
  const float* w_mid = (const float*)d_in[5];
  const float* b_mid = (const float*)d_in[6];
  const float* w_out = (const float*)d_in[7];
  const float* b_out = (const float*)d_in[8];
  const float* e1w = (const float*)d_in[9];
  const float* e1b = (const float*)d_in[10];
  const float* e2w = (const float*)d_in[11];
  const float* e2b = (const float*)d_in[12];

  // workspace carve-out (256B aligned)
  char* wsp = (char*)d_ws;
  size_t off = 0;
  auto alloc = [&](size_t bytes) -> void* {
    void* p = wsp + off;
    off = (off + bytes + 255) & ~(size_t)255;
    return p;
  };
  float* emb1 = (float*)alloc(Bb * Ee * 4);
  float* emb2 = (float*)alloc(Bb * Ee * 4);
  float* x0   = (float*)alloc((size_t)ROWS * Ee * 4);
  float* av   = (float*)alloc((size_t)ROWS * Ee * 4);
  float* x1   = (float*)alloc((size_t)ROWS * Ee * 4);
  float* x2   = (float*)alloc((size_t)ROWS * Ee * 4);
  float* x3   = (float*)alloc((size_t)ROWS * Ee * 4);
  _Float16* h16 = (_Float16*)alloc((size_t)ROWS * Ee * 2);
  _Float16* g16 = (_Float16*)alloc((size_t)ROWS * Ee * 2);
  _Float16* o16 = (_Float16*)alloc((size_t)ROWS * Ee * 2);
  _Float16* r16 = (_Float16*)alloc((size_t)ROWS * Ee * 2);
  size_t qkbytes = (size_t)Bb * Hh * Nn * 32 * 2;  // padded K=32 layout
  _Float16* q16 = (_Float16*)alloc(qkbytes);
  _Float16* k16 = (_Float16*)alloc(qkbytes);
  _Float16* v16 = (_Float16*)alloc((size_t)Bb * Hh * Nn * 16 * 2);

  const dim3 gB(ROWS / 128), bB(256);               // gemm
  const dim3 gL(ROWS / 4),   bL(128);               // layernorm
  const dim3 gA(Bb * Hh * (Nn / 16) / 4), bA(128);  // attention

  auto sa_block = [&](const float* xin, float* xout, bool relu16, _Float16* r16out, int base) {
    const float* ln_g  = (const float*)d_in[base + 0];
    const float* ln_b  = (const float*)d_in[base + 1];
    const float* qkv_w = (const float*)d_in[base + 2];
    const float* qkv_b = (const float*)d_in[base + 3];
    const float* outw  = (const float*)d_in[base + 4];
    const float* outb  = (const float*)d_in[base + 5];
    const float* fln_g = (const float*)d_in[base + 6];
    const float* fln_b = (const float*)d_in[base + 7];
    const float* f1w   = (const float*)d_in[base + 8];
    const float* f1b   = (const float*)d_in[base + 9];
    const float* f2w   = (const float*)d_in[base + 10];
    const float* f2b   = (const float*)d_in[base + 11];

    hipMemsetAsync(q16, 0, qkbytes, stream);  // keep d>=16 pad at zero
    hipMemsetAsync(k16, 0, qkbytes, stream);
    ln_kernel<<<gL, bL, 0, stream>>>(xin, ln_g, ln_b, h16);
    // QKV: <gelu,bvec,resid,storef,s16,qkv>
    gemm_wmma_kernel<0, 0, 0, 0, 0, 1><<<gB, bB, 0, stream>>>(
        h16, qkv_w, qkv_b, nullptr, nullptr, nullptr, nullptr, q16, k16, v16, 192);
    attn_kernel<<<gA, bA, 0, stream>>>(q16, k16, v16, o16);
    // out-proj + residual -> av (f32)
    gemm_wmma_kernel<0, 0, 1, 1, 0, 0><<<gB, bB, 0, stream>>>(
        o16, outw, outb, av, nullptr, xin, nullptr, nullptr, nullptr, nullptr, 64);
    ln_kernel<<<gL, bL, 0, stream>>>(av, fln_g, fln_b, h16);
    // f1 + exact GELU -> fp16
    gemm_wmma_kernel<1, 0, 0, 0, 1, 0><<<gB, bB, 0, stream>>>(
        h16, f1w, f1b, nullptr, g16, nullptr, nullptr, nullptr, nullptr, nullptr, 64);
    // f2 + residual -> f32 (and optionally relu'd fp16 for the next GEMM)
    if (relu16)
      gemm_wmma_kernel<0, 0, 1, 1, 2, 0><<<gB, bB, 0, stream>>>(
          g16, f2w, f2b, xout, r16out, av, nullptr, nullptr, nullptr, nullptr, 64);
    else
      gemm_wmma_kernel<0, 0, 1, 1, 0, 0><<<gB, bB, 0, stream>>>(
          g16, f2w, f2b, xout, nullptr, av, nullptr, nullptr, nullptr, nullptr, 64);
  };

  embed_kernel<<<dim3(Bb), dim3(Ee), 0, stream>>>(t_int, e1w, e1b, e2w, e2b, emb1, emb2);
  input_proj_kernel<<<dim3(ROWS * Ee / 256), dim3(256), 0, stream>>>(pos, w_in, b_in, emb1, x0);

  sa_block(x0, x1, true, r16, 13);  // SA1; also emit relu(x1) fp16 for mid GEMM

  // mid: relu(x1) @ w_mid^T + b_mid + silu-embedding broadcast
  gemm_wmma_kernel<0, 1, 0, 1, 0, 0><<<gB, bB, 0, stream>>>(
      r16, w_mid, b_mid, x2, nullptr, nullptr, emb2, nullptr, nullptr, nullptr, 64);

  sa_block(x2, x3, false, nullptr, 25);  // SA2

  out_proj_kernel<<<dim3((ROWS * 3 + 255) / 256), dim3(256), 0, stream>>>(
      x3, w_out, b_out, node_mask, (float*)d_out);
}
